// ParallelGatedScan_42606075576705
// MI455X (gfx1250) — matrix operand order repaired and checked by reading, not compiled, not run on hardware
//
#include <hip/hip_runtime.h>
#include <hip/hip_bf16.h>

// ---------------------------------------------------------------------------
// Types for WMMA
// ---------------------------------------------------------------------------
typedef __attribute__((ext_vector_type(16))) __bf16 v16bf;
typedef __attribute__((ext_vector_type(8)))  __bf16 v8bf;
typedef __attribute__((ext_vector_type(8)))  float  v8f;

union BF16Frag { v16bf v; v8bf h[2]; };

static __device__ inline __bf16 f32_to_bf16_rne(float f) {
    unsigned int u = __float_as_uint(f);
    u += 0x7fffu + ((u >> 16) & 1u);           // round-to-nearest-even
    unsigned short s = (unsigned short)(u >> 16);
    __bf16 r;
    __builtin_memcpy(&r, &s, 2);
    return r;
}

// Problem constants (match reference setup_inputs)
#define BB 8
#define TT 2048
#define DD 512
#define SD 256
#define OD 64
#define CHUNK 32
#define NCHUNK (TT / CHUNK)      // 64
#define MM (BB * TT)             // 16384

// ---------------------------------------------------------------------------
// f32 -> bf16 row-major convert (for activations)
// ---------------------------------------------------------------------------
__global__ void convert_bf16_kernel(const float* __restrict__ in,
                                    __bf16* __restrict__ out, int n) {
    int i = blockIdx.x * blockDim.x + threadIdx.x;
    if (i < n) out[i] = f32_to_bf16_rne(in[i]);
}

// ---------------------------------------------------------------------------
// Pack weight W[K,N] (f32, row-major) into bf16 fragment layout:
//   Wp[kb32][n][h][j] = bf16( W[kb32*32 + h*16 + j, n] ),  j in 0..15, h in 0..1
// so a B fragment (32x16 tile, column n, lane-half h) is 2 contiguous b128 loads.
// ---------------------------------------------------------------------------
__global__ void pack_weight_kernel(const float* __restrict__ W,
                                   __bf16* __restrict__ Wp, int K, int N) {
    int idx = blockIdx.x * blockDim.x + threadIdx.x;
    int total = K * N;
    if (idx >= total) return;
    int j    = idx & 15;
    int h    = (idx >> 4) & 1;
    int n    = (idx >> 5) % N;
    int kb32 = idx / (N * 32);
    int k = kb32 * 32 + h * 16 + j;
    Wp[idx] = f32_to_bf16_rne(W[(size_t)k * N + n]);
}

// ---------------------------------------------------------------------------
// bf16 WMMA GEMM:  C[M,N] (f32) = A[M,K](bf16 row-major) * Wp(packed) + bias
// 8 waves / 256-thread block; each wave computes a 32x64 C tile via
// 8 accumulators and v_wmma_f32_16x16x32_bf16.
// ---------------------------------------------------------------------------
__global__ __launch_bounds__(256) void gemm_bf16_wmma_kernel(
    const __bf16* __restrict__ A,
    const __bf16* __restrict__ Wp,
    const float*  __restrict__ bias,
    float* __restrict__ C,
    int M, int N, int K, int accumulate)
{
    const int wave  = (int)((blockIdx.x * blockDim.x + threadIdx.x) >> 5);
    const int lane  = threadIdx.x & 31;
    const int tiles_n = N >> 6;               // 64-wide N tiles
    const int wm = wave / tiles_n;
    const int wn = wave % tiles_n;
    const int m0 = wm << 5;                   // 32 rows per wave
    const int n0 = wn << 6;                   // 64 cols per wave
    const int lm = lane & 15;
    const int lh = lane >> 4;

    v8f zero = {0.f, 0.f, 0.f, 0.f, 0.f, 0.f, 0.f, 0.f};
    v8f acc[2][4];
    #pragma unroll
    for (int p = 0; p < 2; ++p)
        #pragma unroll
        for (int s = 0; s < 4; ++s) acc[p][s] = zero;

    const __bf16* Arow0 = A + (size_t)(m0 + lm) * K;
    const __bf16* Arow1 = A + (size_t)(m0 + 16 + lm) * K;

    for (int kb = 0; kb < K; kb += 32) {
        BF16Frag a0, a1;
        // A 16x32 bf16 fragment, contiguous k order used here (B packed to match): lane-half h holds K = kb + h*16 + [0..15]
        a0.h[0] = *(const v8bf*)(Arow0 + kb + lh * 16);
        a0.h[1] = *(const v8bf*)(Arow0 + kb + lh * 16 + 8);
        a1.h[0] = *(const v8bf*)(Arow1 + kb + lh * 16);
        a1.h[1] = *(const v8bf*)(Arow1 + kb + lh * 16 + 8);
        if (kb + 32 < K) {
            __builtin_prefetch(Arow0 + kb + 32, 0, 3);
            __builtin_prefetch(Arow1 + kb + 32, 0, 3);
        }
        const __bf16* Wb = Wp + (size_t)(kb >> 5) * (size_t)N * 32;
        #pragma unroll
        for (int s = 0; s < 4; ++s) {
            const int n = n0 + s * 16 + lm;
            const __bf16* bp = Wb + ((size_t)n * 2 + lh) * 16;
            BF16Frag bfr;
            bfr.h[0] = *(const v8bf*)(bp);
            bfr.h[1] = *(const v8bf*)(bp + 8);
            acc[0][s] = __builtin_amdgcn_wmma_f32_16x16x32_bf16(
                false, a0.v, false, bfr.v, (short)0, acc[0][s], false, false);
            acc[1][s] = __builtin_amdgcn_wmma_f32_16x16x32_bf16(
                false, a1.v, false, bfr.v, (short)0, acc[1][s], false, false);
        }
    }

    // Epilogue: C/D layout — VGPR r, lane L: m = r + 8*(L>>4), n = L&15
    #pragma unroll
    for (int s = 0; s < 4; ++s) {
        const int n  = n0 + s * 16 + lm;
        const float bv = bias ? bias[n] : 0.0f;
        #pragma unroll
        for (int p = 0; p < 2; ++p) {
            #pragma unroll
            for (int r = 0; r < 8; ++r) {
                const int m = m0 + p * 16 + r + lh * 8;
                const size_t idx = (size_t)m * N + n;
                float val = acc[p][s][r] + bv;
                if (accumulate) val += C[idx];
                C[idx] = val;
            }
        }
    }
}

// ---------------------------------------------------------------------------
// gates = sigmoid(gatepre); drive = (1-gates) * inpre   (in place)
// ---------------------------------------------------------------------------
__global__ void gate_drive_kernel(float* __restrict__ gp,
                                  float* __restrict__ ip, int n) {
    int i = blockIdx.x * blockDim.x + threadIdx.x;
    if (i >= n) return;
    float g = 1.0f / (1.0f + expf(-gp[i]));
    gp[i] = g;
    ip[i] = (1.0f - g) * ip[i];
}

// ---------------------------------------------------------------------------
// Chunked scan, pass 1: per-(b,chunk,sd) chunk summaries (ca_last, wb_last)
// ---------------------------------------------------------------------------
__global__ void scan_summary_kernel(const float* __restrict__ g,
                                    const float* __restrict__ d,
                                    float* __restrict__ ca_last,
                                    float* __restrict__ wb_last) {
    int idx = blockIdx.x * blockDim.x + threadIdx.x;   // (b,c,sd) flat: BB*NCHUNK*SD
    int sd = idx & (SD - 1);
    int c  = (idx >> 8) & (NCHUNK - 1);
    int b  = idx >> 14;
    size_t base = (((size_t)b * TT) + (size_t)c * CHUNK) * SD + sd;
    float la = 0.f, wb = 0.f, ca = 1.f;
    for (int k = 0; k < CHUNK; ++k) {
        float a = g[base + (size_t)k * SD];
        la += logf(fmaxf(a, 1e-6f));
        ca = expf(la);
        wb += d[base + (size_t)k * SD] / fmaxf(ca, 1e-8f);
    }
    ca_last[idx] = ca;
    wb_last[idx] = wb;
}

// ---------------------------------------------------------------------------
// Chunked scan, pass 2: sequential carry over 64 chunks per (b,sd)
// ---------------------------------------------------------------------------
__global__ void scan_carry_kernel(const float* __restrict__ ca_last,
                                  const float* __restrict__ wb_last,
                                  float* __restrict__ h_prev) {
    int idx = blockIdx.x * blockDim.x + threadIdx.x;   // BB*SD threads
    int sd = idx & (SD - 1);
    int b  = idx >> 8;
    float h = 0.f;
    for (int c = 0; c < NCHUNK; ++c) {
        size_t o = ((size_t)(b * NCHUNK + c)) * SD + sd;
        h_prev[o] = h;
        h = ca_last[o] * (h + wb_last[o]);
    }
}

// ---------------------------------------------------------------------------
// Chunked scan, pass 3: states = cum_a * (h_prev + cum_wb); write bf16
// ---------------------------------------------------------------------------
__global__ void scan_states_kernel(const float* __restrict__ g,
                                   const float* __restrict__ d,
                                   const float* __restrict__ h_prev,
                                   __bf16* __restrict__ states) {
    int idx = blockIdx.x * blockDim.x + threadIdx.x;
    int sd = idx & (SD - 1);
    int c  = (idx >> 8) & (NCHUNK - 1);
    int b  = idx >> 14;
    size_t base = (((size_t)b * TT) + (size_t)c * CHUNK) * SD + sd;
    float hp = h_prev[idx];
    float la = 0.f, wb = 0.f;
    for (int k = 0; k < CHUNK; ++k) {
        size_t o = base + (size_t)k * SD;
        float a = g[o];
        la += logf(fmaxf(a, 1e-6f));
        float ca = expf(la);
        wb += d[o] / fmaxf(ca, 1e-8f);
        states[o] = f32_to_bf16_rne(ca * (hp + wb));
    }
}

// ---------------------------------------------------------------------------
// Binding scan (mode='outer'): S_t = diag(dec) S_{t-1} + v_t k_t^T; o_t = S_t q_t
// One block per batch; S(64x64) in registers (16 floats/thread), k/q in LDS.
// ---------------------------------------------------------------------------
__global__ __launch_bounds__(256) void binding_scan_kernel(
    const float* __restrict__ v, const float* __restrict__ k,
    const float* __restrict__ q, const float* __restrict__ op_decay,
    __bf16* __restrict__ o)
{
    __shared__ float lk[OD];
    __shared__ float lq[OD];
    __shared__ float opart[256];
    const int tid = threadIdx.x;
    const int b   = blockIdx.x;
    const int i   = tid & 63;      // state row
    const int jg  = tid >> 6;      // column group (4 groups of 16)
    const float dec = 1.0f / (1.0f + expf(-op_decay[i]));

    float S[16];
    #pragma unroll
    for (int jj = 0; jj < 16; ++jj) S[jj] = 0.f;

    for (int t = 0; t < TT; ++t) {
        const size_t row = ((size_t)b * TT + t) * OD;
        if (tid < OD) { lk[tid] = k[row + tid]; lq[tid] = q[row + tid]; }
        __syncthreads();
        const float vi = v[row + i];
        float acc = 0.f;
        #pragma unroll
        for (int jj = 0; jj < 16; ++jj) {
            const int j = jg * 16 + jj;
            S[jj] = dec * S[jj] + vi * lk[j];
            acc += S[jj] * lq[j];
        }
        opart[tid] = acc;          // tid == jg*64 + i
        __syncthreads();
        if (tid < OD) {
            float oo = opart[tid] + opart[64 + tid] + opart[128 + tid] + opart[192 + tid];
            o[row + tid] = f32_to_bf16_rne(oo);
        }
        // next iteration's barriers protect lk/lq/opart reuse
    }
}

// ---------------------------------------------------------------------------
// Final: y = x + (scan_out + bind); LayerNorm over D=512; out f32
// ---------------------------------------------------------------------------
__global__ __launch_bounds__(256) void add_ln_kernel(
    const float* __restrict__ x, const float* __restrict__ ybuf,
    const float* __restrict__ g, const float* __restrict__ beta,
    float* __restrict__ out)
{
    __shared__ float ssum[256];
    __shared__ float ssq[256];
    const int row = blockIdx.x;
    const int tid = threadIdx.x;
    const size_t base = (size_t)row * DD;
    const float y0 = x[base + tid]       + ybuf[base + tid];
    const float y1 = x[base + 256 + tid] + ybuf[base + 256 + tid];
    ssum[tid] = y0 + y1;
    ssq[tid]  = y0 * y0 + y1 * y1;
    __syncthreads();
    for (int s = 128; s > 0; s >>= 1) {
        if (tid < s) { ssum[tid] += ssum[tid + s]; ssq[tid] += ssq[tid + s]; }
        __syncthreads();
    }
    const float mu  = ssum[0] * (1.0f / DD);
    const float var = ssq[0] * (1.0f / DD) - mu * mu;
    const float rs  = rsqrtf(var + 1e-5f);
    out[base + tid]       = (y0 - mu) * rs * g[tid]       + beta[tid];
    out[base + 256 + tid] = (y1 - mu) * rs * g[256 + tid] + beta[256 + tid];
}

// ---------------------------------------------------------------------------
// Host launcher
// ---------------------------------------------------------------------------
static void launch_gemm(const __bf16* A, const __bf16* Wp, const float* bias,
                        float* C, int M, int N, int K, int acc, hipStream_t s) {
    int waves = (M / 32) * (N / 64);
    int blocks = waves / 8;
    gemm_bf16_wmma_kernel<<<blocks, 256, 0, s>>>(A, Wp, bias, C, M, N, K, acc);
}

extern "C" void kernel_launch(void* const* d_in, const int* in_sizes, int n_in,
                              void* d_out, int out_size, void* d_ws, size_t ws_size,
                              hipStream_t stream) {
    const float* x       = (const float*)d_in[0];
    const float* gate_W  = (const float*)d_in[1];
    const float* gate_b  = (const float*)d_in[2];
    const float* in_W    = (const float*)d_in[3];
    const float* in_b    = (const float*)d_in[4];
    const float* out_W   = (const float*)d_in[5];
    const float* out_b   = (const float*)d_in[6];
    const float* opv_W   = (const float*)d_in[7];
    const float* opv_b   = (const float*)d_in[8];
    const float* opk_W   = (const float*)d_in[9];
    const float* opk_b   = (const float*)d_in[10];
    const float* opq_W   = (const float*)d_in[11];
    const float* opq_b   = (const float*)d_in[12];
    const float* op_decay= (const float*)d_in[13];
    const float* opout_W = (const float*)d_in[14];
    const float* opout_b = (const float*)d_in[15];
    const float* ln_g    = (const float*)d_in[16];
    const float* ln_b    = (const float*)d_in[17];
    float* out = (float*)d_out;

    // Workspace carving (256B aligned slices)
    char* p = (char*)d_ws;
    auto alloc = [&](size_t bytes) -> void* {
        void* r = (void*)p;
        p += (bytes + 255) & ~(size_t)255;
        return r;
    };
    __bf16* xb       = (__bf16*)alloc((size_t)MM * DD * 2);
    __bf16* Wp_gate  = (__bf16*)alloc((size_t)DD * SD * 2);
    __bf16* Wp_in    = (__bf16*)alloc((size_t)DD * SD * 2);
    __bf16* Wp_v     = (__bf16*)alloc((size_t)DD * OD * 2);
    __bf16* Wp_k     = (__bf16*)alloc((size_t)DD * OD * 2);
    __bf16* Wp_q     = (__bf16*)alloc((size_t)DD * OD * 2);
    __bf16* Wp_out   = (__bf16*)alloc((size_t)SD * DD * 2);
    __bf16* Wp_opout = (__bf16*)alloc((size_t)OD * DD * 2);
    float*  gatepre  = (float*)alloc((size_t)MM * SD * 4);   // -> gates (in place)
    float*  inpre    = (float*)alloc((size_t)MM * SD * 4);   // -> drive (in place)
    float*  ca_last  = (float*)alloc((size_t)BB * NCHUNK * SD * 4);
    float*  wb_last  = (float*)alloc((size_t)BB * NCHUNK * SD * 4);
    float*  h_prev   = (float*)alloc((size_t)BB * NCHUNK * SD * 4);
    __bf16* states   = (__bf16*)alloc((size_t)MM * SD * 2);
    float*  vbuf     = (float*)alloc((size_t)MM * OD * 4);
    float*  kbuf     = (float*)alloc((size_t)MM * OD * 4);
    float*  qbuf     = (float*)alloc((size_t)MM * OD * 4);
    __bf16* obuf     = (__bf16*)alloc((size_t)MM * OD * 2);
    float*  ybuf     = (float*)alloc((size_t)MM * DD * 4);   // scan_out + bind

    // 1) Convert activations & pack weights to bf16
    {
        int n = MM * DD;
        convert_bf16_kernel<<<n / 256, 256, 0, stream>>>(x, xb, n);
    }
    pack_weight_kernel<<<(DD * SD) / 256, 256, 0, stream>>>(gate_W,  Wp_gate,  DD, SD);
    pack_weight_kernel<<<(DD * SD) / 256, 256, 0, stream>>>(in_W,    Wp_in,    DD, SD);
    pack_weight_kernel<<<(DD * OD) / 256, 256, 0, stream>>>(opv_W,   Wp_v,     DD, OD);
    pack_weight_kernel<<<(DD * OD) / 256, 256, 0, stream>>>(opk_W,   Wp_k,     DD, OD);
    pack_weight_kernel<<<(DD * OD) / 256, 256, 0, stream>>>(opq_W,   Wp_q,     DD, OD);
    pack_weight_kernel<<<(SD * DD) / 256, 256, 0, stream>>>(out_W,   Wp_out,   SD, DD);
    pack_weight_kernel<<<(OD * DD) / 256, 256, 0, stream>>>(opout_W, Wp_opout, OD, DD);

    // 2) Projection GEMMs (WMMA bf16 -> f32)
    launch_gemm(xb, Wp_gate, gate_b, gatepre, MM, SD, DD, 0, stream);
    launch_gemm(xb, Wp_in,   in_b,   inpre,   MM, SD, DD, 0, stream);
    launch_gemm(xb, Wp_v,    opv_b,  vbuf,    MM, OD, DD, 0, stream);
    launch_gemm(xb, Wp_k,    opk_b,  kbuf,    MM, OD, DD, 0, stream);
    launch_gemm(xb, Wp_q,    opq_b,  qbuf,    MM, OD, DD, 0, stream);

    // 3) gates/drive elementwise
    {
        int n = MM * SD;
        gate_drive_kernel<<<n / 256, 256, 0, stream>>>(gatepre, inpre, n);
    }

    // 4) Chunked scan (3 passes), states written as bf16
    scan_summary_kernel<<<(BB * NCHUNK * SD) / 256, 256, 0, stream>>>(gatepre, inpre, ca_last, wb_last);
    scan_carry_kernel<<<(BB * SD) / 256, 256, 0, stream>>>(ca_last, wb_last, h_prev);
    scan_states_kernel<<<(BB * NCHUNK * SD) / 256, 256, 0, stream>>>(gatepre, inpre, h_prev, states);

    // 5) scan_out = states @ out_W + out_b   (into ybuf)
    launch_gemm(states, Wp_out, out_b, ybuf, MM, DD, SD, 0, stream);

    // 6) Binding scan -> o (bf16)
    binding_scan_kernel<<<BB, 256, 0, stream>>>(vbuf, kbuf, qbuf, op_decay, obuf);

    // 7) ybuf += o @ opout_W + opout_b   (accumulate)
    launch_gemm(obuf, Wp_opout, opout_b, ybuf, MM, DD, OD, 1, stream);

    // 8) y = x + ybuf; LayerNorm
    add_ln_kernel<<<MM, 256, 0, stream>>>(x, ybuf, ln_g, ln_b, out);
}
